// Model_Recursive_LSTM_v2_5265629905387
// MI455X (gfx1250) — compile-verified, matrix operand order checked by reading, not run on hardware
//
#include <hip/hip_runtime.h>
#include <hip/hip_bf16.h>

// ---------------------------------------------------------------------------
// CDNA5 (gfx1250) implementation of Model_Recursive_LSTM_v2 forward pass.
// All matmuls -> v_wmma_f32_16x16x32_bf16 (bf16 inputs, fp32 accumulate).
// v3: 128x64 block tile (4 WMMA accumulators per wave sharing one A fragment)
// + double-buffered LDS A-staging (one barrier per K-chunk, next chunk's
// global loads issued before the current WMMAs). Guard-free steady state:
// weights pre-padded (N->64, K->32), activations stored with K-padded leading
// dims (pad cols zero-filled by the epilogue), M always a multiple of 128.
// ---------------------------------------------------------------------------

typedef __bf16 bf16_t;
typedef __attribute__((ext_vector_type(16))) __bf16 v16bf;
typedef __attribute__((ext_vector_type(8)))  __bf16 v8bf;
typedef __attribute__((ext_vector_type(8)))  float  v8f;

#define MBLK 128   // M rows per block (8 waves x 16); all M are multiples of 128
#define NBLK 64    // N cols per block (4 x 16 subtiles per wave)
#define KC   32    // K chunk per WMMA
#define LDSP 40    // LDS row stride in bf16 (80B -> 16B-aligned 8-elem slices)

// C[m,n] = act( A0[m,:]*W0[n,:] + A1[m,:]*W1[n,:] + bias[n] ), n < N
// C[m,n] = 0 for N <= n < Nstore   (zero-fill K-padding for the next layer)
// W is (Nalloc>=gridx*64, K) row-major zero-padded; K multiple of 32.
__global__ __launch_bounds__(256) void gemm_wmma_bf16(
    const float* __restrict__ A0, long long ldA0, long long zsA0,
    const bf16_t* __restrict__ W0, int K0,
    const float* __restrict__ A1, long long ldA1, long long zsA1,
    const bf16_t* __restrict__ W1, int K1,
    const float* __restrict__ bias,
    float* __restrict__ C, long long ldC, long long zsC,
    int N, int Nstore, int act)
{
    __shared__ __align__(16) bf16_t As[2][MBLK][LDSP];
    const int tid  = threadIdx.x;
    const int wave = tid >> 5;          // 0..7 (wave32)
    const int lane = tid & 31;
    const int lo   = lane & 15;
    const int hi   = lane >> 4;
    const int mBase = blockIdx.y * MBLK;
    const int nBase = blockIdx.x * NBLK;
    const int z = blockIdx.z;
    A0 += (long long)z * zsA0;
    if (A1) A1 += (long long)z * zsA1;
    C  += (long long)z * zsC;

    // staging map: thread t covers row (t>>1), 16 consecutive cols at (t&1)*16
    const int sR = tid >> 1;
    const int sC = (tid & 1) * 16;
    const int rowL = wave * 16 + lo;

    v8f acc[4] = {{}, {}, {}, {}};

    for (int term = 0; term < 2; ++term) {
        const float*    A   = term ? A1 : A0;
        const bf16_t*   W   = term ? W1 : W0;
        const int       K   = term ? K1 : K0;
        const long long ldA = term ? ldA1 : ldA0;
        if (A == nullptr) continue;
        const float* Arow = A + (long long)(mBase + sR) * ldA + sC;
        const int nch = K / KC;

        // prologue: stage chunk 0 into buffer 0
        {
            const float4* src = (const float4*)Arow;
            const float4 f0 = src[0], f1 = src[1], f2 = src[2], f3 = src[3];
            v8bf p0, p1;
            p0[0]=(bf16_t)f0.x; p0[1]=(bf16_t)f0.y; p0[2]=(bf16_t)f0.z; p0[3]=(bf16_t)f0.w;
            p0[4]=(bf16_t)f1.x; p0[5]=(bf16_t)f1.y; p0[6]=(bf16_t)f1.z; p0[7]=(bf16_t)f1.w;
            p1[0]=(bf16_t)f2.x; p1[1]=(bf16_t)f2.y; p1[2]=(bf16_t)f2.z; p1[3]=(bf16_t)f2.w;
            p1[4]=(bf16_t)f3.x; p1[5]=(bf16_t)f3.y; p1[6]=(bf16_t)f3.z; p1[7]=(bf16_t)f3.w;
            *(v8bf*)&As[0][sR][sC]     = p0;
            *(v8bf*)&As[0][sR][sC + 8] = p1;
        }
        __syncthreads();

        for (int i = 0; i < nch; ++i) {
            const int cur = i & 1;
            const int k0  = i * KC;
            const bool hasNext = (i + 1) < nch;

            // issue next chunk's global loads early (overlap with WMMAs)
            float4 f0, f1, f2, f3;
            if (hasNext) {
                const float4* src = (const float4*)(Arow + (k0 + KC));
                f0 = src[0]; f1 = src[1]; f2 = src[2]; f3 = src[3];
                __builtin_prefetch(W + (long long)(nBase + lo) * K + k0 + KC, 0, 1);
            }

            // A fragment from current LDS buffer (ISA 16-bit lane swizzle:
            // element j holds K = k0+8*hi+j (j<8) | k0+16+8*hi+(j-8))
            const v8bf a0 = *(const v8bf*)&As[cur][rowL][8 * hi];
            const v8bf a1 = *(const v8bf*)&As[cur][rowL][16 + 8 * hi];
            v16bf av;
            #pragma unroll
            for (int j = 0; j < 8; ++j) { av[j] = a0[j]; av[j + 8] = a1[j]; }

            // 4 N-subtiles reuse the same A fragment
            #pragma unroll
            for (int s = 0; s < 4; ++s) {
                const bf16_t* Wr = W + (long long)(nBase + s * 16 + lo) * K + k0;
                const v8bf b0 = *(const v8bf*)(Wr + 8 * hi);
                const v8bf b1 = *(const v8bf*)(Wr + 16 + 8 * hi);
                v16bf bv;
                #pragma unroll
                for (int j = 0; j < 8; ++j) { bv[j] = b0[j]; bv[j + 8] = b1[j]; }
                acc[s] = __builtin_amdgcn_wmma_f32_16x16x32_bf16(
                    false, av, false, bv, (short)0, acc[s], false, false);
            }

            // convert + store next chunk into the other LDS buffer
            if (hasNext) {
                v8bf p0, p1;
                p0[0]=(bf16_t)f0.x; p0[1]=(bf16_t)f0.y; p0[2]=(bf16_t)f0.z; p0[3]=(bf16_t)f0.w;
                p0[4]=(bf16_t)f1.x; p0[5]=(bf16_t)f1.y; p0[6]=(bf16_t)f1.z; p0[7]=(bf16_t)f1.w;
                p1[0]=(bf16_t)f2.x; p1[1]=(bf16_t)f2.y; p1[2]=(bf16_t)f2.z; p1[3]=(bf16_t)f2.w;
                p1[4]=(bf16_t)f3.x; p1[5]=(bf16_t)f3.y; p1[6]=(bf16_t)f3.z; p1[7]=(bf16_t)f3.w;
                *(v8bf*)&As[1 - cur][sR][sC]     = p0;
                *(v8bf*)&As[1 - cur][sR][sC + 8] = p1;
            }
            __syncthreads();
        }
    }

    // epilogue: element r of acc[s] is row (8*hi + r), col (nBase+16*s+lo)
    #pragma unroll
    for (int s = 0; s < 4; ++s) {
        const int n = nBase + s * 16 + lo;
        if (n < Nstore) {
            const float bv = (n < N && bias) ? bias[n] : 0.0f;
            #pragma unroll
            for (int r = 0; r < 8; ++r) {
                const int m = mBase + wave * 16 + hi * 8 + r;
                float v = acc[s][r] + bv;
                if (act == 1) v = (v > 0.0f) ? v : (__expf(v) - 1.0f);  // ELU
                if (n >= N) v = 0.0f;                                   // K-pad zeros
                C[(long long)m * ldC + n] = v;
            }
        }
    }
}

// LSTM gate update: g=(Z,Bn,gLd) pre-act [i|f|g|o] (4*H<=gLd); h,c=(Z,Bn,hLd)
__global__ void lstm_gates_kernel(const float* __restrict__ g,
                                  float* __restrict__ h, float* __restrict__ c,
                                  int Z, int Bn, int H, int hLd, int gLd)
{
    const long long idx = (long long)blockIdx.x * blockDim.x + threadIdx.x;
    const long long total = (long long)Z * Bn * H;
    if (idx >= total) return;
    const int e = (int)(idx % H);
    const int b = (int)((idx / H) % Bn);
    const int z = (int)(idx / ((long long)H * Bn));
    const float* gr = g + ((long long)z * Bn + b) * gLd;
    const float gi = gr[e], gf = gr[H + e], gc = gr[2 * H + e], go = gr[3 * H + e];
    const float si = 1.0f / (1.0f + __expf(-gi));
    const float sf = 1.0f / (1.0f + __expf(-gf));
    const float so = 1.0f / (1.0f + __expf(-go));
    const float tc = tanhf(gc);
    const long long hi_ = ((long long)z * Bn + b) * hLd + e;
    const float cn = sf * c[hi_] + si * tc;
    c[hi_] = cn;
    h[hi_] = so * tanhf(cn);
}

// dst=(Z,512,384): [0:180)=nodes_h, [180:360)=comps_h, [360:384)=loops[b,loop0+z*step,:]
// per-z h sources have row stride 192.
__global__ void build_cc_in_kernel(float* __restrict__ dst,
    const float* __restrict__ nodes_h, int nodes_bcast,
    const float* __restrict__ comps_h, int comps_bcast,
    const float* __restrict__ loops, int loop0, int loopStep, int Z)
{
    const long long idx = (long long)blockIdx.x * blockDim.x + threadIdx.x;
    const long long total = (long long)Z * 512 * 384;
    if (idx >= total) return;
    const int col = (int)(idx % 384);
    const int b   = (int)((idx / 384) % 512);
    const int z   = (int)(idx / (384LL * 512));
    float v;
    if (col < 180) {
        v = nodes_bcast ? nodes_h[col] : nodes_h[((long long)z * 512 + b) * 192 + col];
    } else if (col < 360) {
        const int e = col - 180;
        v = comps_bcast ? comps_h[e] : comps_h[((long long)z * 512 + b) * 192 + e];
    } else {
        const int f  = col - 360;
        const int li = loop0 + z * loopStep;
        v = loops[((long long)b * 7 + li) * 24 + f];
    }
    dst[idx] = v;
}

// zero-padded bf16 repack: dst(Nalloc,Kpad) <- src(N,K)
__global__ void pad_convert_kernel(const float* __restrict__ s, bf16_t* __restrict__ d,
                                   int N, int K, int Nalloc, int Kpad)
{
    const long long i = (long long)blockIdx.x * blockDim.x + threadIdx.x;
    const long long total = (long long)Nalloc * Kpad;
    if (i >= total) return;
    const int k  = (int)(i % Kpad);
    const int nn = (int)(i / Kpad);
    const float v = (nn < N && k < K) ? s[(long long)nn * K + k] : 0.0f;
    d[i] = (bf16_t)v;
}

__global__ void vadd_kernel(const float* __restrict__ a, const float* __restrict__ b,
                            float* __restrict__ d, int n)
{
    const int i = blockIdx.x * blockDim.x + threadIdx.x;
    if (i < n) d[i] = a[i] + b[i];
}

__global__ void zero_kernel(float* __restrict__ p, long long n)
{
    const long long i = (long long)blockIdx.x * blockDim.x + threadIdx.x;
    if (i < n) p[i] = 0.0f;
}

// ---------------------------------------------------------------------------
// Host orchestration
// ---------------------------------------------------------------------------
static inline void G(hipStream_t s,
                     const float* A0, long long ldA0, long long zsA0, const bf16_t* W0, int K0,
                     const float* A1, long long ldA1, long long zsA1, const bf16_t* W1, int K1,
                     const float* bias, float* C, long long ldC, long long zsC,
                     int M, int N, int Nstore, int act, int Z)
{
    dim3 grid((Nstore + NBLK - 1) / NBLK, M / MBLK, Z);
    gemm_wmma_bf16<<<grid, 256, 0, s>>>(A0, ldA0, zsA0, W0, K0,
                                        A1, ldA1, zsA1, W1, K1,
                                        bias, C, ldC, zsC, N, Nstore, act);
}

extern "C" void kernel_launch(void* const* d_in, const int* in_sizes, int n_in,
                              void* d_out, int out_size, void* d_ws, size_t ws_size,
                              hipStream_t stream)
{
    (void)in_sizes; (void)n_in; (void)out_size; (void)ws_size;
    const int E = 180, EP = 192, G4 = 720, GP = 768, M_CE = 512 * 64;

    // Input order: comps_tensor, loops_tensor, then params flattened as a jax
    // pytree (dict keys sorted): cc, ce, comps_lstm{Whh,Wih,bhh,bih},
    // no_comps, no_nodes, nodes_lstm{...}, pred, reg.
    const float* comps   = (const float*)d_in[0];   // (512,64,1024)
    const float* loops   = (const float*)d_in[1];   // (512,7,24)
    const float* cc0_W   = (const float*)d_in[2];   // (200,384)
    const float* cc0_b   = (const float*)d_in[3];
    const float* cc1_W   = (const float*)d_in[4];   // (180,200)
    const float* cc1_b   = (const float*)d_in[5];
    const float* ce0_W   = (const float*)d_in[6];   // (600,1024)
    const float* ce0_b   = (const float*)d_in[7];
    const float* ce1_W   = (const float*)d_in[8];   // (350,600)
    const float* ce1_b   = (const float*)d_in[9];
    const float* ce2_W   = (const float*)d_in[10];  // (200,350)
    const float* ce2_b   = (const float*)d_in[11];
    const float* ce3_W   = (const float*)d_in[12];  // (180,200)
    const float* ce3_b   = (const float*)d_in[13];
    const float* cWhh    = (const float*)d_in[14];  // (720,180)
    const float* cWih    = (const float*)d_in[15];  // (720,180)
    const float* cbhh    = (const float*)d_in[16];
    const float* cbih    = (const float*)d_in[17];
    const float* no_comps= (const float*)d_in[18];  // (1,180)
    const float* no_nodes= (const float*)d_in[19];  // (1,180)
    const float* nWhh    = (const float*)d_in[20];
    const float* nWih    = (const float*)d_in[21];
    const float* nbhh    = (const float*)d_in[22];
    const float* nbih    = (const float*)d_in[23];
    const float* pred_W  = (const float*)d_in[24];  // (1,180)
    const float* pred_b  = (const float*)d_in[25];
    const float* reg0_W  = (const float*)d_in[26];  // (200,180)
    const float* reg0_b  = (const float*)d_in[27];
    const float* reg1_W  = (const float*)d_in[28];  // (180,200)
    const float* reg1_b  = (const float*)d_in[29];

    // --- workspace carve-out ---
    char* ws = (char*)d_ws;
    size_t off = 0;
    auto alloc = [&](size_t bytes) -> char* {
        char* p = ws + off;
        off += (bytes + 255) & ~(size_t)255;
        return p;
    };
    // padded bf16 weights: (Nalloc (64-mult), Kpad (32-mult))
    bf16_t* w_ce0  = (bf16_t*)alloc((size_t)640 * 1024 * 2);
    bf16_t* w_ce1  = (bf16_t*)alloc((size_t)384 * 640 * 2);
    bf16_t* w_ce2  = (bf16_t*)alloc((size_t)256 * 384 * 2);
    bf16_t* w_ce3  = (bf16_t*)alloc((size_t)192 * 256 * 2);
    bf16_t* w_cc0  = (bf16_t*)alloc((size_t)256 * 384 * 2);
    bf16_t* w_cc1  = (bf16_t*)alloc((size_t)192 * 256 * 2);
    bf16_t* w_rg0  = (bf16_t*)alloc((size_t)256 * 192 * 2);
    bf16_t* w_rg1  = (bf16_t*)alloc((size_t)192 * 256 * 2);
    bf16_t* w_prd  = (bf16_t*)alloc((size_t)64 * 192 * 2);
    bf16_t* w_cWih = (bf16_t*)alloc((size_t)768 * 192 * 2);
    bf16_t* w_cWhh = (bf16_t*)alloc((size_t)768 * 192 * 2);
    bf16_t* w_nWih = (bf16_t*)alloc((size_t)768 * 192 * 2);
    bf16_t* w_nWhh = (bf16_t*)alloc((size_t)768 * 192 * 2);
    float* cb_c = (float*)alloc(720 * 4);
    float* cb_n = (float*)alloc(720 * 4);
    // K-padded fp32 activations
    float* P0   = (float*)alloc((size_t)M_CE * 640 * 4);
    float* P1   = (float*)alloc((size_t)M_CE * 384 * 4);
    float* emb  = (float*)alloc((size_t)M_CE * EP * 4);   // (512,64,192)
    float* h_lf = (float*)alloc((size_t)4 * 512 * EP * 4);
    float* c_lf = (float*)alloc((size_t)4 * 512 * EP * 4);
    float* g_lf = (float*)alloc((size_t)4 * 512 * GP * 4);
    float* ccin = (float*)alloc((size_t)4 * 512 * 384 * 4);
    float* ccmd = (float*)alloc((size_t)4 * 512 * 256 * 4);
    float* lfo  = (float*)alloc((size_t)4 * 512 * EP * 4);
    float* h_nd = (float*)alloc((size_t)2 * 512 * EP * 4);
    float* c_nd = (float*)alloc((size_t)2 * 512 * EP * 4);
    float* g_nd = (float*)alloc((size_t)2 * 512 * GP * 4);
    float* ndin = (float*)alloc((size_t)2 * 512 * 384 * 4);
    float* ndmd = (float*)alloc((size_t)2 * 512 * 256 * 4);
    float* ndo  = (float*)alloc((size_t)2 * 512 * EP * 4);
    float* h_rt = (float*)alloc((size_t)512 * EP * 4);
    float* c_rt = (float*)alloc((size_t)512 * EP * 4);
    float* g_rt = (float*)alloc((size_t)512 * GP * 4);
    float* rtin = (float*)alloc((size_t)512 * 384 * 4);
    float* rtmd = (float*)alloc((size_t)512 * 256 * 4);
    float* rto  = (float*)alloc((size_t)512 * EP * 4);
    float* rgmd = (float*)alloc((size_t)512 * 256 * 4);
    float* rgo  = (float*)alloc((size_t)512 * EP * 4);

    auto pc = [&](const float* s, bf16_t* d, int N, int K, int Na, int Kp) {
        const long long n = (long long)Na * Kp;
        pad_convert_kernel<<<(unsigned)((n + 255) / 256), 256, 0, stream>>>(s, d, N, K, Na, Kp);
    };
    auto zero = [&](float* p, long long n) {
        zero_kernel<<<(unsigned)((n + 255) / 256), 256, 0, stream>>>(p, n);
    };

    // weight repack (fp32 -> zero-padded bf16) + combined LSTM biases
    pc(ce0_W, w_ce0, 600, 1024, 640, 1024);
    pc(ce1_W, w_ce1, 350, 600, 384, 640);
    pc(ce2_W, w_ce2, 200, 350, 256, 384);
    pc(ce3_W, w_ce3, 180, 200, 192, 256);
    pc(cc0_W, w_cc0, 200, 384, 256, 384);
    pc(cc1_W, w_cc1, 180, 200, 192, 256);
    pc(reg0_W, w_rg0, 200, 180, 256, 192);
    pc(reg1_W, w_rg1, 180, 200, 192, 256);
    pc(pred_W, w_prd, 1, 180, 64, 192);
    pc(cWih, w_cWih, 720, 180, 768, 192);
    pc(cWhh, w_cWhh, 720, 180, 768, 192);
    pc(nWih, w_nWih, 720, 180, 768, 192);
    pc(nWhh, w_nWhh, 720, 180, 768, 192);
    vadd_kernel<<<3, 256, 0, stream>>>(cbih, cbhh, cb_c, 720);
    vadd_kernel<<<3, 256, 0, stream>>>(nbih, nbhh, cb_n, 720);

    // --- comps embedding MLP (dominant GEMM chain, ELU) ---
    G(stream, comps, 1024, 0, w_ce0, 1024, nullptr,0,0,nullptr,0, ce0_b, P0, 640, 0, M_CE, 600, 640, 1, 1);
    G(stream, P0,    640,  0, w_ce1, 640,  nullptr,0,0,nullptr,0, ce1_b, P1, 384, 0, M_CE, 350, 384, 1, 1);
    G(stream, P1,    384,  0, w_ce2, 384,  nullptr,0,0,nullptr,0, ce2_b, P0, 256, 0, M_CE, 200, 256, 1, 1);
    G(stream, P0,    256,  0, w_ce3, 256,  nullptr,0,0,nullptr,0, ce3_b, emb, EP, 0, M_CE, 180, EP, 1, 1);

    // --- 4 leaf LSTMs, seq=16, batched over blockIdx.z ---
    zero(h_lf, 4 * 512 * EP); zero(c_lf, 4 * 512 * EP);
    for (int t = 0; t < 16; ++t) {
        // g = x_t @ Wih^T + h @ Whh^T + (bih+bhh); x_t row b = emb[b, 16z+t, :]
        G(stream, emb + (long long)t * EP, 64LL * EP, 16LL * EP, w_cWih, EP,
                  h_lf, EP, 512LL * EP, w_cWhh, EP,
                  cb_c, g_lf, GP, 512LL * GP, 512, G4, GP, 0, 4);
        lstm_gates_kernel<<<(4 * 512 * E + 255) / 256, 256, 0, stream>>>(
            g_lf, h_lf, c_lf, 4, 512, E, EP, GP);
    }
    // leaf cc: concat(no_nodes, comps_h, loops[:,3+z,:]) -> 200 -> 180 (ELU)
    build_cc_in_kernel<<<(4 * 512 * 384 + 255) / 256, 256, 0, stream>>>(
        ccin, no_nodes, 1, h_lf, 0, loops, 3, 1, 4);
    G(stream, ccin, 384, 0, w_cc0, 384, nullptr,0,0,nullptr,0, cc0_b, ccmd, 256, 0, 4 * 512, 200, 256, 1, 1);
    G(stream, ccmd, 256, 0, w_cc1, 256, nullptr,0,0,nullptr,0, cc1_b, lfo,  EP,  0, 4 * 512, 180, EP,  1, 1);

    // --- 2 internal-node LSTMs, seq=2 (children = leaf pairs) ---
    zero(h_nd, 2 * 512 * EP); zero(c_nd, 2 * 512 * EP);
    for (int t = 0; t < 2; ++t) {
        G(stream, lfo + (long long)t * 512 * EP, EP, 2LL * 512 * EP, w_nWih, EP,
                  h_nd, EP, 512LL * EP, w_nWhh, EP,
                  cb_n, g_nd, GP, 512LL * GP, 512, G4, GP, 0, 2);
        lstm_gates_kernel<<<(2 * 512 * E + 255) / 256, 256, 0, stream>>>(
            g_nd, h_nd, c_nd, 2, 512, E, EP, GP);
    }
    build_cc_in_kernel<<<(2 * 512 * 384 + 255) / 256, 256, 0, stream>>>(
        ndin, h_nd, 0, no_comps, 1, loops, 1, 1, 2);
    G(stream, ndin, 384, 0, w_cc0, 384, nullptr,0,0,nullptr,0, cc0_b, ndmd, 256, 0, 2 * 512, 200, 256, 1, 1);
    G(stream, ndmd, 256, 0, w_cc1, 256, nullptr,0,0,nullptr,0, cc1_b, ndo,  EP,  0, 2 * 512, 180, EP,  1, 1);

    // --- root LSTM, seq=2 (children = the two node outputs) ---
    zero(h_rt, 512 * EP); zero(c_rt, 512 * EP);
    for (int t = 0; t < 2; ++t) {
        G(stream, ndo + (long long)t * 512 * EP, EP, 0, w_nWih, EP,
                  h_rt, EP, 0, w_nWhh, EP,
                  cb_n, g_rt, GP, 0, 512, G4, GP, 0, 1);
        lstm_gates_kernel<<<(512 * E + 255) / 256, 256, 0, stream>>>(
            g_rt, h_rt, c_rt, 1, 512, E, EP, GP);
    }
    build_cc_in_kernel<<<(512 * 384 + 255) / 256, 256, 0, stream>>>(
        rtin, h_rt, 0, no_comps, 1, loops, 0, 0, 1);
    G(stream, rtin, 384, 0, w_cc0, 384, nullptr,0,0,nullptr,0, cc0_b, rtmd, 256, 0, 512, 200, 256, 1, 1);
    G(stream, rtmd, 256, 0, w_cc1, 256, nullptr,0,0,nullptr,0, cc1_b, rto,  EP,  0, 512, 180, EP,  1, 1);

    // --- regression head: reg MLP (ELU) then pred (linear, N=1) ---
    G(stream, rto,  EP,  0, w_rg0, EP,  nullptr,0,0,nullptr,0, reg0_b, rgmd, 256, 0, 512, 200, 256, 1, 1);
    G(stream, rgmd, 256, 0, w_rg1, 256, nullptr,0,0,nullptr,0, reg1_b, rgo,  EP,  0, 512, 180, EP,  1, 1);
    G(stream, rgo,  EP,  0, w_prd, EP,  nullptr,0,0,nullptr,0, pred_b, (float*)d_out, 1, 0, 512, 1, 1, 0, 1);
}
